// GNN_52226802319978
// MI455X (gfx1250) — compile-verified, matrix operand order checked by reading, not run on hardware
//
#include <hip/hip_runtime.h>
#include <hip/hip_bf16.h>
#include <stdint.h>

// ---------------------------------------------------------------------------
// Problem constants (match reference)
// ---------------------------------------------------------------------------
#define GN   20000        // nodes
#define GE   320000       // edges (before self loops)
#define GETOT (GE + GN)   // with self loops
#define GH   4            // heads
#define GC   64           // channels per head
#define GF   256          // H*C

// GEMM epilogue flags
#define GEMM_BIAS  1
#define GEMM_RELU  2
#define GEMM_ACCUM 4

typedef __attribute__((ext_vector_type(16))) __bf16 v16bf;
typedef __attribute__((ext_vector_type(8)))  float  v8f;
typedef __attribute__((ext_vector_type(4)))  unsigned v4u;
typedef __attribute__((ext_vector_type(4)))  int      v4i;
typedef __attribute__((ext_vector_type(8)))  int      v8i;

#if defined(__gfx1250__) && __has_builtin(__builtin_amdgcn_tensor_load_to_lds)
#define HAVE_TDM 1
#else
#define HAVE_TDM 0
#endif

// ---------------------------------------------------------------------------
// f32 -> bf16 conversion (activations) ; n multiple of 4
// ---------------------------------------------------------------------------
__global__ void f32_to_bf16_kernel(const float* __restrict__ in,
                                   __bf16* __restrict__ out, long n)
{
    long t = ((long)blockIdx.x * blockDim.x + threadIdx.x) * 4;
    if (t >= n) return;
    float4 v = *reinterpret_cast<const float4*>(in + t);
    out[t + 0] = (__bf16)v.x;
    out[t + 1] = (__bf16)v.y;
    out[t + 2] = (__bf16)v.z;
    out[t + 3] = (__bf16)v.w;
}

// weight transpose+convert:  in[K,Nc] f32  ->  out[Nc,K] bf16
__global__ void w_transpose_bf16_kernel(const float* __restrict__ in,
                                        __bf16* __restrict__ out, int K, int Nc)
{
    int t = blockIdx.x * blockDim.x + threadIdx.x;
    if (t >= K * Nc) return;
    int n = t % Nc, k = t / Nc;
    out[(long)n * K + k] = (__bf16)in[t];
}

// ---------------------------------------------------------------------------
// WMMA bf16 GEMM:  C[M,Nc] = A[M,K] * Bt[Nc,K]^T  (+bias, +relu, +accum)
// A: bf16 row-major [M,K], Bt: bf16 [Nc, ldbt] pre-transposed.
// 128 threads = 4 waves; block tile 64x64, K-step 32.
// Tiles staged by Tensor Data Mover with double-buffered LDS pipelining
// (TENSORcnt-based), else cooperative 128-bit copies.
// ---------------------------------------------------------------------------
#define BM 64
#define BN 64
#define BK 32

#if HAVE_TDM
// Issue one TDM 2D-tile load: BK (dim0, contiguous) x 64 rows of bf16,
// row stride `ld` elements; rows_valid limits dim1 (OOB rows -> zeros).
__device__ __forceinline__ void tdm_load_tile(const __bf16* gptr, unsigned lds_off,
                                              int rows_valid, int ld)
{
    unsigned long long ga = (unsigned long long)(uintptr_t)gptr;
    v4u g0;
    g0[0] = 1u;                                    // count=1 user descriptor
    g0[1] = lds_off;                               // lds_addr (bytes)
    g0[2] = (unsigned)ga;                          // global_addr[31:0]
    g0[3] = (unsigned)((ga >> 32) & 0x01FFFFFFu)   // global_addr[56:32]
            | (2u << 30);                          // type=2 ("image")
    v8i g1;
    g1[0] = 0x10000;                               // data_size=1 (2 bytes)
    g1[1] = (int)(((unsigned)BK) << 16);           // tensor_dim0 = 32
    g1[2] = (int)(((unsigned)rows_valid) << 16);   // tensor_dim1 = rows_valid
    g1[3] = (int)(((unsigned)BK) << 16);           // tile_dim0 = 32
    g1[4] = 64;                                    // tile_dim1 = 64 rows
    g1[5] = ld;                                    // tensor_dim0_stride (elems)
    g1[6] = 0;
    g1[7] = 0;
    v4i gz4 = {0, 0, 0, 0};
#if __clang_major__ >= 23
    v8i gz8 = {0, 0, 0, 0, 0, 0, 0, 0};
    __builtin_amdgcn_tensor_load_to_lds(g0, g1, gz4, gz4, gz8, 0);
#else
    __builtin_amdgcn_tensor_load_to_lds(g0, g1, gz4, gz4, 0);
#endif
}
#endif

__global__ __launch_bounds__(128) void gemm_wmma_bf16(
    const __bf16* __restrict__ A, const __bf16* __restrict__ Bt,
    const float* __restrict__ bias, float* __restrict__ C,
    int M, int K, int Nc, int lda, int ldbt, int ldc, int flags)
{
    __shared__ __attribute__((aligned(16))) __bf16 As[2][BM][BK]; // 2 x 4 KB
    __shared__ __attribute__((aligned(16))) __bf16 Bs[2][BN][BK]; // 2 x 4 KB [n][k]

    const int tid  = threadIdx.x;
    const int wave = tid >> 5;
    const int lane = tid & 31;
    const int m_base = blockIdx.y * BM;
    const int n_base = blockIdx.x * BN;
    const int fr   = lane & 15;   // fragment row (A) / col (B)
    const int kgrp = lane >> 4;   // K half selector per 16-bit A-layout

    const int mrows = (M - m_base < BM) ? (M - m_base) : BM;

    v8f acc[4];
    for (int t = 0; t < 4; ++t)
        for (int j = 0; j < 8; ++j) acc[t][j] = 0.0f;

#if HAVE_TDM
    auto issue = [&](int kk, int buf) {
        if (tid == 0) {
            tdm_load_tile(A  + (long)m_base * lda  + kk,
                          (unsigned)(uintptr_t)&As[buf][0][0], mrows, lda);
            tdm_load_tile(Bt + (long)n_base * ldbt + kk,
                          (unsigned)(uintptr_t)&Bs[buf][0][0], BN, ldbt);
        }
    };
#else
    auto issue = [&](int kk, int buf) {
        for (int idx = tid; idx < (BM * BK) / 8; idx += 128) {
            int r = idx >> 2, q = idx & 3;
            int gr = m_base + r; if (gr >= M) gr = M - 1;   // masked in epilogue
            *reinterpret_cast<uint4*>(&As[buf][r][q * 8]) =
                *reinterpret_cast<const uint4*>(A + (long)gr * lda + kk + q * 8);
        }
        for (int idx = tid; idx < (BN * BK) / 8; idx += 128) {
            int r = idx >> 2, q = idx & 3;
            *reinterpret_cast<uint4*>(&Bs[buf][r][q * 8]) =
                *reinterpret_cast<const uint4*>(Bt + (long)(n_base + r) * ldbt + kk + q * 8);
        }
    };
#endif

    issue(0, 0);  // prologue: tile 0 -> buffer 0

    for (int k0 = 0, it = 0; k0 < K; k0 += BK, ++it) {
        const int cur = it & 1;
        const bool more = (k0 + BK) < K;

        if (more) issue(k0 + BK, cur ^ 1);   // overlap next tile's DMA with compute

#if HAVE_TDM
        // In-order TDM completion: <=2 outstanding means tile `it` is resident.
        if (more) __builtin_amdgcn_s_wait_tensorcnt(2);
        else      __builtin_amdgcn_s_wait_tensorcnt(0);
#endif
        __syncthreads();

        // ---- fragments from LDS: two 128-bit reads each (16-bit A layout) ----
        union { uint4 u[2]; v16bf v; } af, bf[4];
        const int arow = wave * 16 + fr;
        af.u[0] = *reinterpret_cast<const uint4*>(&As[cur][arow][kgrp * 8]);
        af.u[1] = *reinterpret_cast<const uint4*>(&As[cur][arow][16 + kgrp * 8]);
        for (int nt = 0; nt < 4; ++nt) {
            const int bcol = nt * 16 + fr;
            bf[nt].u[0] = *reinterpret_cast<const uint4*>(&Bs[cur][bcol][kgrp * 8]);
            bf[nt].u[1] = *reinterpret_cast<const uint4*>(&Bs[cur][bcol][16 + kgrp * 8]);
        }
        for (int nt = 0; nt < 4; ++nt) {
            acc[nt] = __builtin_amdgcn_wmma_f32_16x16x32_bf16(
                false, af.v, false, bf[nt].v, (short)0, acc[nt], false, false);
        }
        __syncthreads();
    }

    // ---- epilogue: C/D 16x16 f32 layout: vgpr r, lanes 0-15 -> M=r; 16-31 -> M=8+r
    const int mrow0 = m_base + wave * 16 + 8 * (lane >> 4);
    for (int nt = 0; nt < 4; ++nt) {
        const int col = n_base + nt * 16 + (lane & 15);
        for (int r = 0; r < 8; ++r) {
            const int row = mrow0 + r;
            if (row < M && col < Nc) {
                float v = acc[nt][r];
                long off = (long)row * ldc + col;
                if (flags & GEMM_ACCUM) v += C[off];
                if ((flags & GEMM_BIAS) && bias) v += bias[col];
                if (flags & GEMM_RELU) v = v > 0.0f ? v : 0.0f;
                C[off] = v;
            }
        }
    }
}

// ---------------------------------------------------------------------------
// GAT support kernels
// ---------------------------------------------------------------------------
__device__ __forceinline__ unsigned fkey(float f) {
    unsigned u = __float_as_uint(f);
    return (u & 0x80000000u) ? ~u : (u | 0x80000000u);
}
__device__ __forceinline__ float fdecode(unsigned k) {
    return (k & 0x80000000u) ? __uint_as_float(k ^ 0x80000000u)
                             : __uint_as_float(~k);
}

// al[n,h] = <h[n,h,:], a_src[h,:]> ; ar likewise
__global__ void attn_logits_kernel(const float* __restrict__ hbuf,
                                   const float* __restrict__ a_src,
                                   const float* __restrict__ a_dst,
                                   float* __restrict__ al, float* __restrict__ ar)
{
    int t = blockIdx.x * blockDim.x + threadIdx.x;
    if (t >= GN * GH) return;
    int h = t & (GH - 1);
    const float* hp = hbuf + (long)(t >> 2) * GF + h * GC;
    const float* as = a_src + h * GC;
    const float* ad = a_dst + h * GC;
    float sl = 0.0f, sr = 0.0f;
    for (int c = 0; c < GC; ++c) { float hv = hp[c]; sl += hv * as[c]; sr += hv * ad[c]; }
    al[t] = sl; ar[t] = sr;
}

// agg <- bias (broadcast), mkey <- 0 (== -inf key), ssum <- 0
__global__ void layer_init_kernel(float* __restrict__ agg, const float* __restrict__ bias,
                                  unsigned* __restrict__ mkey, float* __restrict__ ssum)
{
    int t = blockIdx.x * blockDim.x + threadIdx.x;
    if (t >= GN * GF) return;
    agg[t] = bias[t & (GF - 1)];
    if (t < GN * GH) { mkey[t] = 0u; ssum[t] = 0.0f; }
}

// e = leaky_relu(al[src]+ar[dst]); segment max into mkey[dst]
__global__ void edge_pass1_kernel(const int* __restrict__ src_e, const int* __restrict__ dst_e,
                                  const float* __restrict__ al, const float* __restrict__ ar,
                                  float* __restrict__ ebuf, unsigned* __restrict__ mkey)
{
    int t = blockIdx.x * blockDim.x + threadIdx.x;
    if (t >= GETOT * GH) return;
    int e = t >> 2, h = t & (GH - 1);
    int s = (e < GE) ? src_e[e] : (e - GE);
    int d = (e < GE) ? dst_e[e] : (e - GE);
    float v = al[s * GH + h] + ar[d * GH + h];
    v = v > 0.0f ? v : 0.2f * v;
    ebuf[t] = v;
    atomicMax(&mkey[d * GH + h], fkey(v));
}

// ex = exp(e - m[dst]); segment sum into ssum[dst]; ebuf overwritten with ex
__global__ void edge_pass2_kernel(const int* __restrict__ src_e, const int* __restrict__ dst_e,
                                  float* __restrict__ ebuf, const unsigned* __restrict__ mkey,
                                  float* __restrict__ ssum)
{
    int t = blockIdx.x * blockDim.x + threadIdx.x;
    if (t >= GETOT * GH) return;
    int e = t >> 2, h = t & (GH - 1);
    int d = (e < GE) ? dst_e[e] : (e - GE);
    float ex = __expf(ebuf[t] - fdecode(mkey[d * GH + h]));
    ebuf[t] = ex;
    atomicAdd(&ssum[d * GH + h], ex);
}

// one block per edge: agg[dst, :] += alpha[h] * h[src, :]
__global__ __launch_bounds__(256) void edge_pass3_kernel(
    const int* __restrict__ src_e, const int* __restrict__ dst_e,
    const float* __restrict__ ebuf, const float* __restrict__ ssum,
    const float* __restrict__ hbuf, float* __restrict__ agg)
{
    int e = blockIdx.x;
    int t = threadIdx.x;                 // 0..255  (h = t/64, c = t%64)
    int s = (e < GE) ? src_e[e] : (e - GE);
    int d = (e < GE) ? dst_e[e] : (e - GE);
    int h = t >> 6;
    float alpha = ebuf[e * GH + h] / (ssum[d * GH + h] + 1e-16f);
    atomicAdd(&agg[(long)d * GF + t], alpha * hbuf[(long)s * GF + t]);
}

__global__ void relu_kernel(float* __restrict__ z, int n)
{
    int t = blockIdx.x * blockDim.x + threadIdx.x;
    if (t < n) { float v = z[t]; z[t] = v > 0.0f ? v : 0.0f; }
}

// d_out = [ b_sel (N*128) | t_sel (N*128) | gnn (N*64, written by fc3 GEMM) | batch (N) ]
__global__ void final_gather_kernel(const int* __restrict__ voc,
                                    const float* __restrict__ bil,
                                    const float* __restrict__ trf,
                                    const int* __restrict__ batch,
                                    float* __restrict__ out)
{
    int t = blockIdx.x * blockDim.x + threadIdx.x;
    const int R = GN * 128;
    if (t < R) {
        int n = t >> 7, j = t & 127;
        out[t] = bil[(long)voc[n] * 128 + j];
    } else if (t < 2 * R) {
        int u = t - R, n = u >> 7, j = u & 127;
        out[R + u] = trf[(long)voc[n] * 128 + j];
    } else if (t < 2 * R + GN) {
        int n = t - 2 * R;
        out[2 * R + GN * 64 + n] = (float)batch[n];
    }
}

// ---------------------------------------------------------------------------
// Launch sequence
// ---------------------------------------------------------------------------
extern "C" void kernel_launch(void* const* d_in, const int* in_sizes, int n_in,
                              void* d_out, int out_size, void* d_ws, size_t ws_size,
                              hipStream_t stream)
{
    (void)in_sizes; (void)n_in; (void)out_size; (void)ws_size;

    const float* x     = (const float*)d_in[0];
    const int*   ei    = (const int*)d_in[1];
    const int*   batch = (const int*)d_in[2];
    const int*   voc   = (const int*)d_in[3];
    const float* bil   = (const float*)d_in[4];
    const float* trf   = (const float*)d_in[5];
    const float* W[3]    = { (const float*)d_in[6],  (const float*)d_in[10], (const float*)d_in[14] };
    const float* asrc[3] = { (const float*)d_in[7],  (const float*)d_in[11], (const float*)d_in[15] };
    const float* adst[3] = { (const float*)d_in[8],  (const float*)d_in[12], (const float*)d_in[16] };
    const float* bb[3]   = { (const float*)d_in[9],  (const float*)d_in[13], (const float*)d_in[17] };
    const float* fcmax_w = (const float*)d_in[18];
    const float* fcmax_b = (const float*)d_in[19];
    const float* fc1_w   = (const float*)d_in[20];
    const float* fc1_b   = (const float*)d_in[21];
    const float* fc2_w   = (const float*)d_in[22];
    const float* fc2_b   = (const float*)d_in[23];
    const float* fc3_w   = (const float*)d_in[24];
    const float* fc3_b   = (const float*)d_in[25];

    const int* src_e = ei;           // edge_index[0]
    const int* dst_e = ei + GE;      // edge_index[1]

    // ----- workspace carve -----
    float* p    = (float*)d_ws;
    float* hbuf = p;             p += (long)GN * GF;        // [N,256] h = x@W
    float* xall = p;             p += 6L * GN * GF;         // 6 x [N,256]
    float* al   = p;             p += (long)GN * GH;
    float* ar   = p;             p += (long)GN * GH;
    unsigned* mkey = (unsigned*)p; p += (long)GN * GH;
    float* ssum = p;             p += (long)GN * GH;
    float* ebuf = p;             p += (long)GETOT * GH;
    float* z1   = p;             p += (long)GN * 512;
    float* z2   = p;             p += (long)GN * 128;

    __bf16* bp   = (__bf16*)p;
    __bf16* abf  = bp;  bp += (long)GN * 512;   // A operand staging (bf16)
    __bf16* w0t  = bp;  bp += 128 * 256;        // [256][128]
    __bf16* w1t  = bp;  bp += 256 * 256;        // [256][256]
    __bf16* w2t  = bp;  bp += 256 * 256;
    __bf16* fcmt = bp;  bp += 256 * 256;
    __bf16* fc1t = bp;  bp += 1536 * 512;       // [512][1536]
    __bf16* fc2t = bp;  bp += 512 * 128;        // [128][512]
    __bf16* fc3t = bp;  bp += 128 * 64;         // [64][128]
    const __bf16* wlt[3] = { w0t, w1t, w2t };

    // ----- convert + transpose all weights to bf16 [Nc][K] -----
    auto wconv = [&](const float* w, __bf16* wt, int K, int Nc) {
        w_transpose_bf16_kernel<<<(K * Nc + 255) / 256, 256, 0, stream>>>(w, wt, K, Nc);
    };
    wconv(W[0],    w0t,  128,  256);
    wconv(W[1],    w1t,  256,  256);
    wconv(W[2],    w2t,  256,  256);
    wconv(fcmax_w, fcmt, 256,  256);
    wconv(fc1_w,   fc1t, 1536, 512);
    wconv(fc2_w,   fc2t, 512,  128);
    wconv(fc3_w,   fc3t, 128,  64);

    // ----- GEMM helper: convert A to bf16 then WMMA GEMM -----
    auto gemm = [&](const float* Af32, const __bf16* Bt, const float* bias, float* Cm,
                    int M, int K, int Nc, int ldbt, int ldc, int flags) {
        long n = (long)M * K;
        f32_to_bf16_kernel<<<(int)((n / 4 + 255) / 256), 256, 0, stream>>>(Af32, abf, n);
        dim3 grid((Nc + BN - 1) / BN, (M + BM - 1) / BM);
        gemm_wmma_bf16<<<grid, 128, 0, stream>>>(abf, Bt, bias, Cm, M, K, Nc, K, ldbt, ldc, flags);
    };

    const float* layer_in = x;
    int Kin = 128;
    for (int l = 0; l < 3; ++l) {
        float* agg = xall + (long)(2 * l) * GN * GF;       // GAT output slot
        float* rel = xall + (long)(2 * l + 1) * GN * GF;   // fcmax-relu slot

        // h = layer_in @ W[l]     [N,256]  (WMMA)
        gemm(layer_in, wlt[l], nullptr, hbuf, GN, Kin, GF, Kin, GF, 0);

        // attention logits + edge softmax + scatter-aggregate (+bias preload)
        attn_logits_kernel<<<(GN * GH + 255) / 256, 256, 0, stream>>>(hbuf, asrc[l], adst[l], al, ar);
        layer_init_kernel<<<(GN * GF + 255) / 256, 256, 0, stream>>>(agg, bb[l], mkey, ssum);
        edge_pass1_kernel<<<(GETOT * GH + 255) / 256, 256, 0, stream>>>(src_e, dst_e, al, ar, ebuf, mkey);
        edge_pass2_kernel<<<(GETOT * GH + 255) / 256, 256, 0, stream>>>(src_e, dst_e, ebuf, mkey, ssum);
        edge_pass3_kernel<<<GETOT, 256, 0, stream>>>(src_e, dst_e, ebuf, ssum, hbuf, agg);

        // h = relu(agg @ fcmax_w + fcmax_b)   (WMMA)
        gemm(agg, fcmt, fcmax_b, rel, GN, GF, GF, GF, GF, GEMM_BIAS | GEMM_RELU);

        layer_in = rel;
        Kin = GF;
    }

    // fc1: virtual concat [N,1536] @ [1536,512] as 6 accumulating chunk GEMMs
    for (int c = 0; c < 6; ++c) {
        gemm(xall + (long)c * GN * GF, fc1t + (long)c * GF,
             (c == 0) ? fc1_b : nullptr, z1,
             GN, GF, 512, 1536, 512,
             (c == 0) ? GEMM_BIAS : GEMM_ACCUM);
    }
    relu_kernel<<<(GN * 512 + 255) / 256, 256, 0, stream>>>(z1, GN * 512);

    // fc2: relu([N,512] @ [512,128] + b)
    gemm(z1, fc2t, fc2_b, z2, GN, 512, 128, 512, 128, GEMM_BIAS | GEMM_RELU);

    // fc3: [N,128] @ [128,64] + b  -> directly into gnn_feature region of d_out
    float* gnn = (float*)d_out + 2L * GN * 128;
    gemm(z2, fc3t, fc3_b, gnn, GN, 128, 64, 128, 64, GEMM_BIAS);

    // b_sel / t_sel gathers + batch passthrough
    final_gather_kernel<<<(GN * 257 + 255) / 256, 256, 0, stream>>>(voc, bil, trf, batch, (float*)d_out);
}
